// LocalWindowAttention_29025388986372
// MI455X (gfx1250) — matrix operand order repaired and checked by reading, not compiled, not run on hardware
//
#include <hip/hip_runtime.h>
#include <hip/hip_bf16.h>

// ---------- WMMA types (gfx1250, wave32) ----------
typedef __attribute__((ext_vector_type(16))) __bf16 v16bf;
typedef __attribute__((ext_vector_type(8)))  float  v8f;

union FragB16 {
  v16bf v;
  unsigned short us[16];
  uint4 q[2];
};

__device__ __forceinline__ unsigned short f32_to_bf16(float f) {
  unsigned int u = __float_as_uint(f);
  unsigned int r = u + 0x7FFFu + ((u >> 16) & 1u);   // round-to-nearest-even
  return (unsigned short)(r >> 16);
}
__device__ __forceinline__ float bf16_to_f32(unsigned short h) {
  return __uint_as_float(((unsigned int)h) << 16);
}

// ---------- problem constants ----------
#define BATCH   16
#define GRID_H  64
#define GRID_W  64
#define CDIM    768
#define NHEAD   12
#define HD      64
#define WS      8
#define NWIN    1024              // BATCH * 8 * 8
#define NTOK    65                // 64 image tokens + unseen
#define MROWS   (NWIN * NTOK)     // 66560 = 520 * 128
#define N3C     (3 * CDIM)        // 2304
#define NPAD    80                // token padding for attention (5 x 16)
#define KPAD    96                // K padding for P@V (3 x 32)

// GEMM block tile
#define BM 128
#define BN 64
#define BK 32

// =====================================================================
// 1) fp32 -> bf16 convert (weights)
// =====================================================================
__global__ void cvt_bf16_kernel(const float* __restrict__ in,
                                unsigned short* __restrict__ out, size_t n) {
  size_t i = (size_t)blockIdx.x * blockDim.x + threadIdx.x;
  if (i < n) out[i] = f32_to_bf16(in[i]);
}

// =====================================================================
// 2) window partition + unseen append, fp32 -> bf16
// =====================================================================
__global__ void gather_kernel(const float* __restrict__ x,
                              const float* __restrict__ unseen,
                              unsigned short* __restrict__ xw) {
  size_t idx = (size_t)blockIdx.x * blockDim.x + threadIdx.x;
  const size_t total = (size_t)MROWS * CDIM;
  if (idx >= total) return;
  int c   = (int)(idx % CDIM);
  size_t row = idx / CDIM;
  int tok = (int)(row % NTOK);
  int win = (int)(row / NTOK);
  int b  = win >> 6;
  int wy = (win >> 3) & 7;
  int wx = win & 7;
  float v;
  if (tok == 64) {
    v = unseen[(size_t)b * CDIM + c];
  } else {
    int y = wy * WS + (tok >> 3);
    int xq = wx * WS + (tok & 7);
    v = x[(((size_t)b * (GRID_H * GRID_W)) + (size_t)y * GRID_W + xq) * CDIM + c];
  }
  xw[idx] = f32_to_bf16(v);
}

// =====================================================================
// 3/6) LDS-tiled bf16 WMMA GEMM: Out[M,N] = A[M,K] * W[N,K]^T + bias
//      block = 8 waves -> 128x64 tile; double-buffered LDS, K step 32.
//      grid: (N/64, M/128)
// =====================================================================
template <bool F32OUT>
__global__ __launch_bounds__(256)
void gemm_bf16_kernel(const unsigned short* __restrict__ A,
                      const unsigned short* __restrict__ Wm,
                      const float* __restrict__ bias,
                      unsigned short* __restrict__ OutB,
                      float* __restrict__ OutF,
                      int M, int N, int K) {
  __shared__ unsigned short Ab[2][BM][BK];   // 16 KB
  __shared__ unsigned short Bb[2][BN][BK];   // 8 KB

  const int bn = blockIdx.x, bm = blockIdx.y;
  const int t = threadIdx.x;
  const int wave = t >> 5, lane = t & 31;
  const int r = lane & 15, h = lane >> 4;
  const int steps = K / BK;

  // staging addresses: each thread owns (row = t>>2 [+64 for A], chunk = t&3)
  const int srow = t >> 2, schunk = (t & 3) * 8;
  const unsigned short* pa = A  + (size_t)(bm * BM + srow) * K + schunk;
  const unsigned short* pb = Wm + (size_t)(bn * BN + srow) * K + schunk;

  uint4 ra0, ra1, rb;
  // prologue: fetch tile 0
  ra0 = *(const uint4*)(pa);
  ra1 = *(const uint4*)(pa + (size_t)64 * K);
  rb  = *(const uint4*)(pb);
  *(uint4*)&Ab[0][srow][schunk]      = ra0;
  *(uint4*)&Ab[0][64 + srow][schunk] = ra1;
  *(uint4*)&Bb[0][srow][schunk]      = rb;

  v8f acc[4] = {};
  for (int s = 0; s < steps; ++s) {
    const int k0n = (s + 1) * BK;
    if (s + 1 < steps) {                       // fetch next tile into regs
      ra0 = *(const uint4*)(pa + k0n);
      ra1 = *(const uint4*)(pa + (size_t)64 * K + k0n);
      rb  = *(const uint4*)(pb + k0n);
      if (s + 2 < steps) {                     // L2 prefetch two steps out
        __builtin_prefetch(pa + k0n + BK, 0, 3);
        __builtin_prefetch(pb + k0n + BK, 0, 3);
      }
    }
    __syncthreads();                           // tile s visible in LDS
    const int buf = s & 1;
    FragB16 a;
    a.q[0] = *(const uint4*)&Ab[buf][wave * 16 + r][8 * h];
    a.q[1] = *(const uint4*)&Ab[buf][wave * 16 + r][16 + 8 * h];
#pragma unroll
    for (int tn = 0; tn < 4; ++tn) {
      FragB16 b;
      b.q[0] = *(const uint4*)&Bb[buf][tn * 16 + r][8 * h];
      b.q[1] = *(const uint4*)&Bb[buf][tn * 16 + r][16 + 8 * h];
      acc[tn] = __builtin_amdgcn_wmma_f32_16x16x32_bf16(false, a.v, false, b.v,
                                                        (short)0, acc[tn],
                                                        false, false);
    }
    if (s + 1 < steps) {                       // stage next tile (other buffer)
      const int nb = buf ^ 1;
      *(uint4*)&Ab[nb][srow][schunk]      = ra0;
      *(uint4*)&Ab[nb][64 + srow][schunk] = ra1;
      *(uint4*)&Bb[nb][srow][schunk]      = rb;
    }
  }

  // epilogue: bias + store 16x64 strip
#pragma unroll
  for (int tn = 0; tn < 4; ++tn) {
    int n = bn * BN + tn * 16 + r;
    float bv = bias[n];
#pragma unroll
    for (int i = 0; i < 8; ++i) {
      int m = bm * BM + wave * 16 + i + 8 * h;
      if (F32OUT) OutF[(size_t)m * N + n] = acc[tn][i] + bv;
      else        OutB[(size_t)m * N + n] = f32_to_bf16(acc[tn][i] + bv);
    }
  }
}

// =====================================================================
// 4) RoPE2D on q/k image tokens, pack q/k/v into [win, head, 80, 64] bf16
// =====================================================================
__global__ void rope_pack_kernel(const unsigned short* __restrict__ qkv,
                                 unsigned short* __restrict__ qp,
                                 unsigned short* __restrict__ kp,
                                 unsigned short* __restrict__ vp) {
  size_t idx = (size_t)blockIdx.x * blockDim.x + threadIdx.x;
  const size_t total = (size_t)NWIN * NHEAD * NPAD * HD;
  if (idx >= total) return;
  int d    = (int)(idx & 63);
  int tok  = (int)((idx >> 6) % NPAD);
  int head = (int)(((idx >> 6) / NPAD) % NHEAD);
  int win  = (int)((idx >> 6) / (NPAD * NHEAD));
  if (tok >= NTOK) { qp[idx] = 0; kp[idx] = 0; vp[idx] = 0; return; }

  size_t base = (size_t)(win * NTOK + tok) * N3C + head * HD;
  float qv = bf16_to_f32(qkv[base + d]);
  float kv = bf16_to_f32(qkv[base + CDIM + d]);
  float vv = bf16_to_f32(qkv[base + 2 * CDIM + d]);
  vp[idx] = f32_to_bf16(vv);
  if (tok == 64) { qp[idx] = f32_to_bf16(qv); kp[idx] = f32_to_bf16(kv); return; }

  int half = d >> 5;          // 0 -> y, 1 -> x
  int j    = d & 31;
  int wy = (win >> 3) & 7, wx = win & 7;
  float p = half ? (float)(wx * WS + (tok & 7))
                 : (float)(wy * WS + (tok >> 3));
  int jj = j & 15;
  float theta = p * powf(100.0f, -(float)jj / 16.0f);  // inv_freq, d=32
  float c = cosf(theta), s = sinf(theta);
  int pd = half * 32 + ((j < 16) ? (j + 16) : (j - 16));
  float qpart = bf16_to_f32(qkv[base + pd]);
  float kpart = bf16_to_f32(qkv[base + CDIM + pd]);
  float qo = (j < 16) ? (qv * c - qpart * s) : (qv * c + qpart * s);
  float ko = (j < 16) ? (kv * c - kpart * s) : (kv * c + kpart * s);
  qp[idx] = f32_to_bf16(qo);
  kp[idx] = f32_to_bf16(ko);
}

// =====================================================================
// 5) attention: one block per (window, head); 8 waves, WMMA for QK^T and PV
// =====================================================================
__global__ __launch_bounds__(256)
void attn_kernel(const unsigned short* __restrict__ qp,
                 const unsigned short* __restrict__ kp,
                 const unsigned short* __restrict__ vp,
                 unsigned short* __restrict__ attn_out) {
  __shared__ float          S[NPAD][NPAD];   // 25.6 KB
  __shared__ unsigned short P[NPAD][KPAD];   // 15.4 KB
  __shared__ unsigned short Vt[HD][KPAD];    // 12.3 KB (transposed: K-contig per d)

  int win = blockIdx.x, head = blockIdx.y;
  int tid = threadIdx.x;
  int wave = tid >> 5, lane = tid & 31;
  int r = lane & 15, h = lane >> 4;

  size_t base = (size_t)(win * NHEAD + head) * NPAD * HD;
  const unsigned short* q = qp + base;
  const unsigned short* k = kp + base;
  const unsigned short* v = vp + base;

  // stage V transposed: Vt[d][kk] = v[kk][d]; zero-pad kk in [80,96)
  for (int idx = tid; idx < HD * KPAD; idx += 256) {
    int d = idx / KPAD, kk = idx % KPAD;
    Vt[d][kk] = (kk < NPAD) ? v[kk * HD + d] : (unsigned short)0;
  }

  // ---- S = scale * q @ k^T, masked at col >= 65 ----
  for (int t = wave; t < 25; t += 8) {
    int ti = t / 5, tj = t % 5;
    const unsigned short* arow = q + (ti * 16 + r) * HD;
    const unsigned short* brow = k + (tj * 16 + r) * HD;
    v8f acc = {};
#pragma unroll
    for (int k0 = 0; k0 < HD; k0 += 32) {
      FragB16 a, b;
      a.q[0] = *(const uint4*)(arow + k0 + 8 * h);
      a.q[1] = *(const uint4*)(arow + k0 + 16 + 8 * h);
      b.q[0] = *(const uint4*)(brow + k0 + 8 * h);
      b.q[1] = *(const uint4*)(brow + k0 + 16 + 8 * h);
      acc = __builtin_amdgcn_wmma_f32_16x16x32_bf16(false, a.v, false, b.v,
                                                    (short)0, acc, false, false);
    }
    int col = tj * 16 + r;
#pragma unroll
    for (int i = 0; i < 8; ++i) {
      int row = ti * 16 + i + 8 * h;
      S[row][col] = (col < NTOK) ? acc[i] * 0.125f : -3.0e38f;  // hd^-0.5
    }
  }
  __syncthreads();

  // ---- row softmax -> P (bf16, zero-padded to 96 cols) ----
  if (tid < NPAD) {
    float mx = -3.0e38f;
    for (int c = 0; c < NPAD; ++c) mx = fmaxf(mx, S[tid][c]);
    float sum = 0.f;
    for (int c = 0; c < NPAD; ++c) {
      float e = __expf(S[tid][c] - mx);
      S[tid][c] = e;
      sum += e;
    }
    float inv = 1.0f / fmaxf(sum, 1e-20f);
    for (int c = 0; c < NPAD; ++c) P[tid][c] = f32_to_bf16(S[tid][c] * inv);
    for (int c = NPAD; c < KPAD; ++c) P[tid][c] = 0;
  }
  __syncthreads();

  // ---- O = P @ V ----
  for (int t = wave; t < 20; t += 8) {
    int ti = t / 4, tn = t % 4;
    const unsigned short* prow = &P[ti * 16 + r][0];
    const unsigned short* vrow = &Vt[tn * 16 + r][0];
    v8f acc = {};
#pragma unroll
    for (int k0 = 0; k0 < KPAD; k0 += 32) {
      FragB16 a, b;
      a.q[0] = *(const uint4*)(prow + k0 + 8 * h);
      a.q[1] = *(const uint4*)(prow + k0 + 16 + 8 * h);
      b.q[0] = *(const uint4*)(vrow + k0 + 8 * h);
      b.q[1] = *(const uint4*)(vrow + k0 + 16 + 8 * h);
      acc = __builtin_amdgcn_wmma_f32_16x16x32_bf16(false, a.v, false, b.v,
                                                    (short)0, acc, false, false);
    }
    int d = tn * 16 + r;
#pragma unroll
    for (int i = 0; i < 8; ++i) {
      int tok = ti * 16 + i + 8 * h;
      if (tok < NTOK)
        attn_out[(size_t)(win * NTOK + tok) * CDIM + head * HD + d] =
            f32_to_bf16(acc[i]);
    }
  }
}

// =====================================================================
// 7) window merge (image tokens) + unseen mean over 64 windows
// =====================================================================
__global__ void scatter_img_kernel(const float* __restrict__ po,
                                   float* __restrict__ out) {
  size_t idx = (size_t)blockIdx.x * blockDim.x + threadIdx.x;
  const size_t total = (size_t)BATCH * GRID_H * GRID_W * CDIM;
  if (idx >= total) return;
  int c = (int)(idx % CDIM);
  size_t t = idx / CDIM;
  int flat = (int)(t % (GRID_H * GRID_W));
  int b = (int)(t / (GRID_H * GRID_W));
  int y = flat >> 6, xq = flat & 63;
  int wy = y >> 3, ry = y & 7, wx = xq >> 3, rx = xq & 7;
  int win = b * 64 + wy * 8 + wx;
  int tok = ry * WS + rx;
  out[idx] = po[((size_t)win * NTOK + tok) * CDIM + c];
}

__global__ void unseen_mean_kernel(const float* __restrict__ po,
                                   float* __restrict__ out) {
  int idx = blockIdx.x * blockDim.x + threadIdx.x;
  if (idx >= BATCH * CDIM) return;
  int c = idx % CDIM, b = idx / CDIM;
  float s = 0.f;
  for (int w = 0; w < 64; ++w)
    s += po[((size_t)(b * 64 + w) * NTOK + 64) * CDIM + c];
  out[idx] = s * (1.0f / 64.0f);
}

// =====================================================================
// launch
// =====================================================================
extern "C" void kernel_launch(void* const* d_in, const int* in_sizes, int n_in,
                              void* d_out, int out_size, void* d_ws, size_t ws_size,
                              hipStream_t stream) {
  (void)in_sizes; (void)n_in; (void)out_size; (void)ws_size;
  const float* x_img   = (const float*)d_in[0];
  // d_in[1] = pos2d_img (positions recomputed analytically)
  const float* unseen  = (const float*)d_in[2];
  const float* qkv_w   = (const float*)d_in[3];
  const float* qkv_b   = (const float*)d_in[4];
  const float* proj_w  = (const float*)d_in[5];
  const float* proj_b  = (const float*)d_in[6];
  float* out = (float*)d_out;

  // ---- workspace carve-up (bytes) ----
  char* w = (char*)d_ws;
  auto take = [&](size_t bytes) { char* p = w; w += (bytes + 255) & ~(size_t)255; return p; };
  unsigned short* qkv_w_bf = (unsigned short*)take((size_t)N3C * CDIM * 2);
  unsigned short* proj_w_bf= (unsigned short*)take((size_t)CDIM * CDIM * 2);
  unsigned short* xw       = (unsigned short*)take((size_t)MROWS * CDIM * 2);
  unsigned short* qkvb     = (unsigned short*)take((size_t)MROWS * N3C * 2);
  unsigned short* qp       = (unsigned short*)take((size_t)NWIN * NHEAD * NPAD * HD * 2);
  unsigned short* kp       = (unsigned short*)take((size_t)NWIN * NHEAD * NPAD * HD * 2);
  unsigned short* vp       = (unsigned short*)take((size_t)NWIN * NHEAD * NPAD * HD * 2);
  unsigned short* ao       = (unsigned short*)take((size_t)MROWS * CDIM * 2);
  float*          po       = (float*)take((size_t)MROWS * CDIM * 4);

  const int T = 256;
  auto blocks = [](size_t n, int t) { return (unsigned)((n + t - 1) / t); };

  // 1) weights -> bf16
  cvt_bf16_kernel<<<blocks((size_t)N3C * CDIM, T), T, 0, stream>>>(qkv_w, qkv_w_bf, (size_t)N3C * CDIM);
  cvt_bf16_kernel<<<blocks((size_t)CDIM * CDIM, T), T, 0, stream>>>(proj_w, proj_w_bf, (size_t)CDIM * CDIM);

  // 2) window gather + unseen append
  gather_kernel<<<blocks((size_t)MROWS * CDIM, T), T, 0, stream>>>(x_img, unseen, xw);

  // 3) QKV GEMM (bf16 WMMA, bf16 out): grid (N/64, M/128)
  gemm_bf16_kernel<false><<<dim3(N3C / BN, MROWS / BM), T, 0, stream>>>(
      xw, qkv_w_bf, qkv_b, qkvb, nullptr, MROWS, N3C, CDIM);

  // 4) RoPE2D + pack padded q/k/v
  rope_pack_kernel<<<blocks((size_t)NWIN * NHEAD * NPAD * HD, T), T, 0, stream>>>(qkvb, qp, kp, vp);

  // 5) attention per (window, head)
  attn_kernel<<<dim3(NWIN, NHEAD), T, 0, stream>>>(qp, kp, vp, ao);

  // 6) proj GEMM (bf16 WMMA, f32 out)
  gemm_bf16_kernel<true><<<dim3(CDIM / BN, MROWS / BM), T, 0, stream>>>(
      ao, proj_w_bf, proj_b, nullptr, po, MROWS, CDIM, CDIM);

  // 7) merge windows + unseen mean
  scatter_img_kernel<<<blocks((size_t)BATCH * GRID_H * GRID_W * CDIM, T), T, 0, stream>>>(po, out);
  unseen_mean_kernel<<<blocks((size_t)BATCH * CDIM, T), T, 0, stream>>>(
      po, out + (size_t)BATCH * GRID_H * GRID_W * CDIM);
}
